// RuleMemory_62758062129605
// MI455X (gfx1250) — matrix operand order repaired and checked by reading, not compiled, not run on hardware
//
#include <hip/hip_runtime.h>
#include <math.h>

typedef __bf16 bf16_t;
typedef __attribute__((ext_vector_type(16))) __bf16 v16bf;
typedef __attribute__((ext_vector_type(8)))  float  v8f;

#define Ntok 4096
#define Udim 64
#define Bdim 256
#define Sdim 256
#define Rdim 512
#define KDIM 16384   // U*B
#define EMA  0.99f

// ---------------------------------------------------------------- utilities
__global__ __launch_bounds__(256) void zero_f32(float* __restrict__ p, size_t n) {
    size_t i = (size_t)blockIdx.x * blockDim.x + threadIdx.x;
    size_t stride = (size_t)gridDim.x * blockDim.x;
    for (; i < n; i += stride) p[i] = 0.0f;
}

// Swizzled, valid-masked, bf16 B matrix in WMMA-fragment-major order.
// Element (c, ct, lane, i)  [c = K-chunk of 32, ct = 16-col tile, i = 0..15]
//   holds Bv[k, col] with k = 32c + 16*(lane/16) + i, col = 16*ct + (lane&15),
// laid out flat so each lane's 16 bf16 (32 bytes) are contiguous.
__global__ __launch_bounds__(256)
void prep_Bsw(const float* __restrict__ proto, const float* __restrict__ usage,
              bf16_t* __restrict__ Bsw, int ncols) {
    size_t idx = (size_t)blockIdx.x * 256 + threadIdx.x;
    size_t total = (size_t)KDIM * (size_t)ncols;
    if (idx >= total) return;
    const int ntiles = ncols >> 4;
    int i    = (int)(idx & 15);
    int lane = (int)((idx >> 4) & 31);
    size_t rest = idx >> 9;
    int ct = (int)(rest % (size_t)ntiles);
    int c  = (int)(rest / (size_t)ntiles);
    int k  = 32 * c + ((lane >> 4) << 4) + i;
    int col = (ct << 4) + (lane & 15);
    float v = (usage[k] > 0.0f) ? proto[(size_t)k * ncols + col] : 0.0f;
    Bsw[idx] = (bf16_t)v;
}

// ------------------------------------------------- per-row metadata (one block per n)
__global__ __launch_bounds__(256)
void row_meta(const float* __restrict__ qu, const float* __restrict__ qb,
              const float* __restrict__ usage, const float* __restrict__ ema,
              const unsigned char* __restrict__ wmask,
              float* __restrict__ scale, float* __restrict__ base_raw,
              int* __restrict__ cellIdx, float* __restrict__ wval,
              float* __restrict__ sum_w, float* __restrict__ cnt,
              float* __restrict__ accum) {
    const int n = blockIdx.x;
    const int t = threadIdx.x;
    __shared__ float qu_s[Udim];
    __shared__ float qb_s[Bdim];
    __shared__ float red[256];
    __shared__ int   redi[256];

    if (t < Udim) qu_s[t] = qu[n * Udim + t];
    qb_s[t] = qb[n * Bdim + t];
    __syncthreads();

    // argmax over q_b (256 values)
    red[t] = qb_s[t]; redi[t] = t;
    __syncthreads();
    for (int s = 128; s > 0; s >>= 1) {
        if (t < s && red[t + s] > red[t]) { red[t] = red[t + s]; redi[t] = redi[t + s]; }
        __syncthreads();
    }
    const float bind_conf = red[0];
    const int   bind_id   = redi[0];
    __syncthreads();

    // argmax over q_u (64 values)
    red[t] = (t < Udim) ? qu_s[t] : -1e30f; redi[t] = t;
    __syncthreads();
    for (int s = 128; s > 0; s >>= 1) {
        if (t < s && red[t + s] > red[t]) { red[t] = red[t + s]; redi[t] = redi[t + s]; }
        __syncthreads();
    }
    const float op_conf = red[0];
    const int   op_id   = redi[0];
    __syncthreads();

    // denom[n] and base_raw[n] = sum over k of qu*qb*valid*{1, ema}
    float d = 0.0f, br = 0.0f;
    for (int k = t; k < KDIM; k += 256) {
        float vld = (usage[k] > 0.0f) ? 1.0f : 0.0f;
        float a = qu_s[k >> 8] * qb_s[k & 255] * vld;
        d += a;
        br += a * ema[k];
    }
    red[t] = d; __syncthreads();
    for (int s = 128; s > 0; s >>= 1) { if (t < s) red[t] += red[t + s]; __syncthreads(); }
    d = red[0]; __syncthreads();
    red[t] = br; __syncthreads();
    for (int s = 128; s > 0; s >>= 1) { if (t < s) red[t] += red[t + s]; __syncthreads(); }
    br = red[0];

    if (t == 0) {
        scale[n]    = (d > 0.0f) ? (1.0f / fmaxf(d, 1e-6f)) : 0.0f;
        base_raw[n] = br;
        float mf = wmask[n] ? 1.0f : 0.0f;
        float wc = op_conf * bind_conf * mf;
        int cell = op_id * Bdim + bind_id;
        cellIdx[n] = cell;
        wval[n] = wc;
        atomicAdd(&sum_w[cell], wc);
        atomicAdd(&cnt[cell], mf);
        atomicAdd(&accum[0], mf);   // write_rate numerator
    }
}

// ------------------------------------------------- scatter weighted sums into cells
__global__ __launch_bounds__(256)
void scatter_rows(const float* __restrict__ target, const float* __restrict__ qsig,
                  const int* __restrict__ cellIdx, const float* __restrict__ wval,
                  float* __restrict__ sumd, float* __restrict__ sums) {
    const int n = blockIdx.x;
    const float w = wval[n];
    if (w == 0.0f) return;
    const int cell = cellIdx[n];
    const int t = threadIdx.x;
    for (int r = t; r < Rdim; r += 256)
        atomicAdd(&sumd[(size_t)cell * Rdim + r], w * target[(size_t)n * Rdim + r]);
    atomicAdd(&sums[(size_t)cell * Sdim + t], w * qsig[(size_t)n * Sdim + t]);
}

// ------------------------------------------------- WMMA GEMM
// out[n,c] = scale[n] * sum_k A[n,k] Bv[k,c];  A[n,k] = q_u[n,k>>8]*q_b[n,k&255]
// (valid already folded into Bsw).  Each wave: 16 x (16*NACC) output strip.
// Double-buffered B fragments: loads for chunk c+1 issue before WMMAs of chunk c.
template <int NACC>
__global__ __launch_bounds__(256)
void gemm_retrieve(const float* __restrict__ qu, const float* __restrict__ qb,
                   const bf16_t* __restrict__ Bsw, const float* __restrict__ scale,
                   float* __restrict__ out, int ncols) {
    __shared__ float qu_s[16 * Udim];
    __shared__ float qb_s[16 * Bdim];

    const int ntiles  = ncols >> 4;
    const int rowBase = blockIdx.y * 16;
    const int wave = threadIdx.x >> 5;    // 0..7
    const int lane = threadIdx.x & 31;
    const int m    = lane & 15;           // A row within tile / output col id
    const int hf   = lane >> 4;           // lane half selects K sub-range
    const int ct0  = blockIdx.x * (8 * NACC) + wave * NACC;  // first col tile of wave

    for (int i = threadIdx.x; i < 16 * Udim; i += 256)
        qu_s[i] = qu[(size_t)(rowBase + (i >> 6)) * Udim + (i & 63)];
    for (int i = threadIdx.x; i < 16 * Bdim; i += 256)
        qb_s[i] = qb[(size_t)(rowBase + (i >> 8)) * Bdim + (i & 255)];
    __syncthreads();

    // A fragment builder: element i of lane holds K = 32*c + 16*hf + i
    auto buildA = [&](int c) -> v16bf {
        const int   k0  = c * 32;
        const float quv = qu_s[m * Udim + (k0 >> 8)];
        const int   bb  = k0 & 255;
        v16bf a;
#pragma unroll
        for (int i = 0; i < 16; ++i) {
            int j = (hf << 4) + i;
            a[i] = (bf16_t)(quv * qb_s[m * Bdim + bb + j]);
        }
        return a;
    };

    const size_t chunkStride = (size_t)ntiles * 512;   // elements per K-chunk
    const bf16_t* bp = Bsw + ((size_t)ct0 * 32 + lane) * 16;

    v8f acc[NACC] = {};
    v16bf bEven[NACC], bOdd[NACC];

    // prologue: fragments for chunk 0
#pragma unroll
    for (int q = 0; q < NACC; ++q)
        bEven[q] = *(const v16bf*)(bp + (size_t)q * 512);

    for (int c = 0; c < KDIM / 32; c += 2) {
        const bf16_t* bp1 = bp + chunkStride;
        const bf16_t* bp2 = bp1 + chunkStride;

        // issue loads for chunk c+1, then compute chunk c
#pragma unroll
        for (int q = 0; q < NACC; ++q)
            bOdd[q] = *(const v16bf*)(bp1 + (size_t)q * 512);
        __builtin_prefetch((const void*)bp2, 0, 1);
        {
            v16bf a = buildA(c);
#pragma unroll
            for (int q = 0; q < NACC; ++q)
                acc[q] = __builtin_amdgcn_wmma_f32_16x16x32_bf16(
                    false, a, false, bEven[q], (short)0, acc[q], false, false);
        }

        // issue loads for chunk c+2 (phantom read on last iter stays inside d_ws),
        // then compute chunk c+1
#pragma unroll
        for (int q = 0; q < NACC; ++q)
            bEven[q] = *(const v16bf*)(bp2 + (size_t)q * 512);
        {
            v16bf a = buildA(c + 1);
#pragma unroll
            for (int q = 0; q < NACC; ++q)
                acc[q] = __builtin_amdgcn_wmma_f32_16x16x32_bf16(
                    false, a, false, bOdd[q], (short)0, acc[q], false, false);
        }

        bp = bp2;
    }

    const int col0 = (ct0 << 4) + m;
#pragma unroll
    for (int v = 0; v < 8; ++v) {
        int row = rowBase + v + (hf << 3);      // C/D layout: VGPR v -> M = v + 8*(lane/16)
        float sc = scale[row];
#pragma unroll
        for (int q = 0; q < NACC; ++q)
            out[(size_t)row * (size_t)ncols + col0 + 16 * q] = acc[q][v] * sc;
    }
}

// ------------------------------------------------- confidence (one block per n)
__global__ __launch_bounds__(256)
void conf_kernel(const float* __restrict__ qsig, const float* __restrict__ memsig,
                 const float* __restrict__ base_raw, const float* __restrict__ scale,
                 float* __restrict__ conf_out) {
    const int n = blockIdx.x;
    const int t = threadIdx.x;
    __shared__ float r0[256], r1[256], r2[256];
    float qs = qsig[(size_t)n * Sdim + t];
    float ms = memsig[(size_t)n * Sdim + t] + 1e-6f;
    r0[t] = qs * qs; r1[t] = ms * ms; r2[t] = qs * ms;
    __syncthreads();
    for (int s = 128; s > 0; s >>= 1) {
        if (t < s) { r0[t] += r0[t + s]; r1[t] += r1[t + s]; r2[t] += r2[t + s]; }
        __syncthreads();
    }
    if (t == 0) {
        float nq = fmaxf(sqrtf(r0[0]), 1e-12f);
        float nm = fmaxf(sqrtf(r1[0]), 1e-12f);
        float cosv = r2[0] / (nq * nm);
        float base = base_raw[n] * scale[n];
        float v = base * 0.5f * (1.0f + cosv);
        conf_out[n] = fminf(fmaxf(v, 0.0f), 1.0f);
    }
}

// ------------------------------------------------- in-place EMA finalize (one block per cell)
__global__ __launch_bounds__(256)
void finalize_cells(const float* __restrict__ dproto, const float* __restrict__ sproto,
                    const float* __restrict__ ema, const float* __restrict__ usage,
                    const float* __restrict__ sum_w, const float* __restrict__ cnt_,
                    float* __restrict__ outD, float* __restrict__ outS,
                    float* __restrict__ outE, float* __restrict__ outU,
                    float* __restrict__ accum) {
    const int cell = blockIdx.x;
    const int t = threadIdx.x;
    const float sw = sum_w[cell];
    const float c  = cnt_[cell];
    const bool upd = (c > 0.0f);
    const float cw = fmaxf(sw, 1e-6f);

    for (int r = t; r < Rdim; r += 256) {
        size_t idx = (size_t)cell * Rdim + r;
        float p = dproto[idx];
        float s = outD[idx];                       // accumulated weighted sum
        outD[idx] = upd ? (EMA * p + (1.0f - EMA) * (s / cw)) : p;
    }
    {
        size_t idx = (size_t)cell * Sdim + t;
        float p = sproto[idx];
        float s = outS[idx];
        outS[idx] = upd ? (EMA * p + (1.0f - EMA) * (s / cw)) : p;
    }
    if (t == 0) {
        float cm = sw / fmaxf(c, 1.0f);
        float e = ema[cell];
        outE[cell] = upd ? (EMA * e + (1.0f - EMA) * cm) : e;
        float nu = usage[cell] + c;
        outU[cell] = nu;
        atomicAdd(&accum[1], nu);                        // total usage
        atomicAdd(&accum[2], (nu > 0.0f) ? 1.0f : 0.0f); // occupied cells
    }
}

// ------------------------------------------------- scalar stats (single block)
__global__ __launch_bounds__(256)
void stats_kernel(const float* __restrict__ outU, const float* __restrict__ accum,
                  float* __restrict__ out3) {
    __shared__ float red[256];
    const int t = threadIdx.x;
    const float total = fmaxf(accum[1], 1e-6f);
    float ent = 0.0f;
    for (int c = t; c < KDIM; c += 256) {
        float p = outU[c] / total;
        ent -= p * logf(p + 1e-6f);
    }
    red[t] = ent; __syncthreads();
    for (int s = 128; s > 0; s >>= 1) { if (t < s) red[t] += red[t + s]; __syncthreads(); }
    if (t == 0) {
        out3[0] = accum[0] / (float)Ntok;      // write_rate
        out3[1] = accum[2] / (float)KDIM;      // usage_fraction
        out3[2] = red[0] / logf((float)KDIM);  // usage_entropy (normalized)
    }
}

// ================================================================ launcher
extern "C" void kernel_launch(void* const* d_in, const int* in_sizes, int n_in,
                              void* d_out, int out_size, void* d_ws, size_t ws_size,
                              hipStream_t stream) {
    const float* qu      = (const float*)d_in[0];   // [N,U]
    const float* qb      = (const float*)d_in[1];   // [N,B]
    const float* qsig    = (const float*)d_in[2];   // [N,S]
    const float* target  = (const float*)d_in[3];   // [N,R]
    const unsigned char* wmask = (const unsigned char*)d_in[4]; // [N] bool
    const float* dproto  = (const float*)d_in[5];   // [U,B,R]
    const float* sproto  = (const float*)d_in[6];   // [U,B,S]
    const float* ema     = (const float*)d_in[7];   // [U,B]
    const float* usage   = (const float*)d_in[8];   // [U,B]

    float* out  = (float*)d_out;
    float* memD = out;               // [N,R]   offset 0
    float* memS = out + 2097152;     // [N,S]
    float* memC = out + 3145728;     // [N]
    float* newD = out + 3149824;     // [UB,R]  (used as sum accumulator, then final)
    float* newS = out + 11538432;    // [UB,S]
    float* newE = out + 15732736;    // [UB]
    float* newU = out + 15749120;    // [UB]
    float* out3 = out + 15765504;    // [3]

    char* ws = (char*)d_ws;
    bf16_t* Bd     = (bf16_t*)(ws);              // swizzled, 16384*512*2 = 16 MiB
    bf16_t* Bs     = (bf16_t*)(ws + 16777216);   // swizzled, 16384*256*2 =  8 MiB
    float* scale   = (float*)(ws + 25165824);    // [N]
    float* baseraw = (float*)(ws + 25182208);    // [N]
    int*   cellIdx = (int*)  (ws + 25198592);    // [N]
    float* wval    = (float*)(ws + 25214976);    // [N]
    float* sum_w   = (float*)(ws + 25231360);    // [UB]
    float* cnt     = (float*)(ws + 25296896);    // [UB]  (contiguous after sum_w)
    float* accum   = (float*)(ws + 25362432);    // [4]   (contiguous after cnt)

    // 1) zero accumulation regions (deterministic per call)
    zero_f32<<<2048, 256, 0, stream>>>(newD, (size_t)12582912);          // newD + newS contiguous
    zero_f32<<<64,   256, 0, stream>>>(sum_w, (size_t)(KDIM * 2 + 4));   // sum_w + cnt + accum

    // 2) fold valid mask into protos, convert to bf16, swizzle to fragment order
    prep_Bsw<<<(KDIM * Rdim + 255) / 256, 256, 0, stream>>>(dproto, usage, Bd, Rdim);
    prep_Bsw<<<(KDIM * Sdim + 255) / 256, 256, 0, stream>>>(sproto, usage, Bs, Sdim);

    // 3) per-row denom/base/argmax + cell atomics
    row_meta<<<Ntok, 256, 0, stream>>>(qu, qb, usage, ema, wmask,
                                       scale, baseraw, cellIdx, wval, sum_w, cnt, accum);

    // 4) scatter weighted sums into output proto regions
    scatter_rows<<<Ntok, 256, 0, stream>>>(target, qsig, cellIdx, wval, newD, newS);

    // 5) WMMA GEMMs for retrieval
    //    R: each wave 16x64 strip (4 acc) -> block covers 16 rows x 512 cols
    //    S: each wave 16x32 strip (2 acc) -> block covers 16 rows x 256 cols
    dim3 gD(Rdim / 512, Ntok / 16), gS(Sdim / 256, Ntok / 16);
    gemm_retrieve<4><<<gD, 256, 0, stream>>>(qu, qb, Bd, scale, memD, Rdim);
    gemm_retrieve<2><<<gS, 256, 0, stream>>>(qu, qb, Bs, scale, memS, Sdim);

    // 6) confidence from signature agreement
    conf_kernel<<<Ntok, 256, 0, stream>>>(qsig, memS, baseraw, scale, memC);

    // 7) EMA finalize (in place over accumulated sums) + usage
    finalize_cells<<<KDIM, 256, 0, stream>>>(dproto, sproto, ema, usage,
                                             sum_w, cnt, newD, newS, newE, newU, accum);

    // 8) scalar stats
    stats_kernel<<<1, 256, 0, stream>>>(newU, accum, out3);
}
